// dijetResNetBlock_58763742544550
// MI455X (gfx1250) — matrix operand order repaired and checked by reading, not compiled, not run on hardware
//
#include <hip/hip_runtime.h>
#include <stdint.h>

// dijetResNetBlock on gfx1250: 3x (GEMM M=393216,K=192,N=64) + ReLU + residual.
// bf16 WMMA (f32 accumulate): compute ~6us < HBM roofline ~17us (400MB @ 23.3TB/s),
// so the kernel streams x/d/out exactly once with coalesced 128b (async) traffic.

#define ND 64
#define NPAIR 6
#define BPW 16                    // batches per workgroup
#define ROWS (BPW * NPAIR)        // 96 rows (= 6 wave tiles of 16)
#define NWAVE 6
#define NTHREADS (NWAVE * 32)     // 192
#define KCH 6                     // six K-chunks of 32 (4 from x, 2 from d)

typedef __attribute__((ext_vector_type(16))) __bf16 bf16x16;
typedef __attribute__((ext_vector_type(8)))  float  f32x8;

union Frag {
  bf16x16  v;
  uint32_t w[8];
  uint4    q[2];
};

// pack two f32 -> bf16 pair: one v_add each (round-half-up) + one v_perm_b32
__device__ __forceinline__ uint32_t pk2(float lo, float hi) {
  uint32_t ul = __float_as_uint(lo) + 0x8000u;
  uint32_t uh = __float_as_uint(hi) + 0x8000u;
  return __builtin_amdgcn_perm(uh, ul, 0x07060302u);  // {uh[31:16], ul[31:16]}
}

#if __has_builtin(__builtin_amdgcn_global_load_async_to_lds_b128) && \
    __has_builtin(__builtin_amdgcn_global_store_async_from_lds_b128) && \
    __has_builtin(__builtin_amdgcn_s_wait_asynccnt)
#define USE_ASYNC 1
typedef int v4i __attribute__((vector_size(16)));
typedef __attribute__((address_space(1))) v4i gv4i;   // int4 in global AS
typedef __attribute__((address_space(3))) v4i sv4i;   // int4 in LDS AS
#else
#define USE_ASYNC 0
#endif

__global__ __launch_bounds__(NTHREADS)
void dijet_resnet_wmma(const float* __restrict__ x,
                       const float* __restrict__ d,
                       const float* __restrict__ W1, const float* __restrict__ b1,
                       const float* __restrict__ W2, const float* __restrict__ b2,
                       const float* __restrict__ W3, const float* __restrict__ b3,
                       float* __restrict__ out)
{
  __shared__ uint32_t sB[3 * KCH * 4 * 256];   // 72 KB: bf16-pair B fragments, lane-contiguous
  __shared__ float    sBias[3 * ND];           // 768 B
  __shared__ float    sD[BPW * ND * NPAIR];    // 24 KB: d1 stage (layer-1 A + residual)
  __shared__ uint16_t sCur[ROWS * ND];         // 12 KB: inter-layer activations (bf16)
  __shared__ float    sOut[BPW * ND * NPAIR];  // 24 KB: output stage (global layout)

  const int tid = threadIdx.x;
  const int b0  = blockIdx.x * BPW;

  const float* Wl[3] = {W1, W2, W3};
  const float* bv[3] = {b1, b2, b3};

  // ---- stage d1 tile: async global->LDS, overlapped with weight-fragment build ----
  {
    const float4* g  = (const float4*)(d + (size_t)b0 * ND * NPAIR);
    float4*       s4 = (float4*)sD;
#if USE_ASYNC
    #pragma unroll
    for (int k = 0; k < (BPW * ND * NPAIR / 4) / NTHREADS; ++k) {   // 8
      int i = tid + k * NTHREADS;
      __builtin_amdgcn_global_load_async_to_lds_b128(
          (gv4i*)(g + i), (sv4i*)(s4 + i), 0, 0);
    }
#else
    #pragma unroll
    for (int k = 0; k < (BPW * ND * NPAIR / 4) / NTHREADS; ++k)
      s4[tid + k * NTHREADS] = g[tid + k * NTHREADS];
#endif
  }

  // ---- build weight fragments in LDS (16-bit B frags in contiguous k = 16h + s order, K pairs packed) ----
  for (int k = 0; k < (3 * KCH * 4 * 256) / NTHREADS; ++k) {   // 96 words/thread
    int idx  = tid + k * NTHREADS;
    int f    = idx >> 8;                    // (layer, chunk, ntile)
    int rem  = idx & 255;
    int lane = rem >> 3;
    int j    = rem & 7;                     // B VGPR index
    int l    = f / (KCH * 4);
    int fc   = f - l * (KCH * 4);
    int c    = fc >> 2;
    int t    = fc & 3;
    int lh   = lane >> 4;                   // K-half select (B: lanes 16-31 -> K+16)
    int o    = t * 16 + (lane & 15);        // output channel (N)
    const float* W = Wl[l];
    float v0, v1;
    if (c < 4) {                            // x-part of K: pair = W[o,i,0], W[o,i,1]
      int i = c * 16 + lh * 8 + j;
      v0 = W[o * 192 + i * 3 + 0];
      v1 = W[o * 192 + i * 3 + 1];
    } else {                                // d-part of K: pair = W[o,i,2], W[o,i+1,2]
      int i = (c - 4) * 32 + lh * 16 + 2 * j;
      v0 = W[o * 192 + i * 3 + 2];
      v1 = W[o * 192 + (i + 1) * 3 + 2];
    }
    sB[idx] = pk2(v0, v1);
  }
  sBias[tid] = bv[tid >> 6][tid & 63];      // 192 == 3*ND

#if USE_ASYNC
  __builtin_amdgcn_s_wait_asynccnt(0);
#endif
  __syncthreads();

  const int wave = tid >> 5;
  const int lane = tid & 31;
  const int lh   = lane >> 4;
  const int nl   = lane & 15;
  const int rowB = wave * 16;               // this wave's 16-row M tile

  // ---- gather x A-fragments once; reused by all 3 layers (contiguous k = 16h + s order) ----
  Frag ax[4];
  {
    const int r   = rowB + nl;
    const int blc = r / 6;
    const int s   = r - blc * 6;
    const float* xr = x + (size_t)(b0 + blc) * (ND * 12) + 2 * s;
    #pragma unroll
    for (int c = 0; c < 4; ++c) {
      #pragma unroll
      for (int j = 0; j < 8; ++j) {
        int i = c * 16 + lh * 8 + j;
        float2 p = *(const float2*)(xr + i * 12);       // (k=0,1) contiguous pair
        ax[c].w[j] = pk2(p.x, p.y);
      }
    }
  }

  const int rA  = rowB + nl;
  const int blA = rA / 6;
  const int sA  = rA - blA * 6;

  // layer loop intentionally NOT unrolled: keeps live VGPRs < 256
  for (int l = 0; l < 3; ++l) {
    // A-fragments for the d-part of K (layer 1 from staged d1, else from sCur)
    Frag ad[2];
    #pragma unroll
    for (int c = 0; c < 2; ++c) {
      #pragma unroll
      for (int j = 0; j < 8; ++j) {
        int i = c * 32 + lh * 16 + j * 2;
        if (l == 0) {
          float v0 = sD[blA * 384 + i * 6 + sA];
          float v1 = sD[blA * 384 + (i + 1) * 6 + sA];
          ad[c].w[j] = pk2(v0, v1);
        } else {
          ad[c].w[j] = *(const uint32_t*)&sCur[(rowB + nl) * ND + i];  // (i,i+1) packed
        }
      }
    }

    // one N-tile at a time (rolled): only 8 accumulator VGPRs + one tile's B frags live
    for (int t = 0; t < 4; ++t) {
      const uint32_t* bbase = &sB[(l * KCH * 4 + t) * 256 + lane * 8];
      f32x8 acc = (f32x8){0.f, 0.f, 0.f, 0.f, 0.f, 0.f, 0.f, 0.f};
      #pragma unroll
      for (int c = 0; c < KCH; ++c) {
        Frag bf;
        const uint4* p = (const uint4*)(bbase + c * 1024);
        bf.q[0] = p[0];
        bf.q[1] = p[1];
        if (c < 4) {
          acc = __builtin_amdgcn_wmma_f32_16x16x32_bf16(
              false, ax[c].v, false, bf.v, (short)0, acc, false, false);
        } else {
          acc = __builtin_amdgcn_wmma_f32_16x16x32_bf16(
              false, ad[c - 4].v, false, bf.v, (short)0, acc, false, false);
        }
      }
      const float bias = sBias[l * ND + t * 16 + nl];
      if (l < 2) {
        // +bias, ReLU -> bf16 -> LDS for next layer's A d-fragments
        #pragma unroll
        for (int v = 0; v < 8; ++v) {
          int m = v + 8 * lh;
          float r = fmaxf(acc[v] + bias, 0.f);
          sCur[(rowB + m) * ND + t * 16 + nl] = (uint16_t)pk2(r, r);
        }
      } else {
        // +bias, +d1 residual, final ReLU -> output stage (global [b][o][s] layout)
        #pragma unroll
        for (int v = 0; v < 8; ++v) {
          int m  = v + 8 * lh;
          int rr = rowB + m;
          int bb = rr / 6;
          int ss = rr - bb * 6;
          int o  = t * 16 + nl;
          sOut[bb * 384 + o * 6 + ss] = fmaxf(acc[v] + bias + sD[bb * 384 + o * 6 + ss], 0.f);
        }
      }
    }
    __syncthreads();
  }

  // ---- coalesced 128b output store (async LDS->global when available) ----
  {
    float4*       g  = (float4*)(out + (size_t)b0 * ND * NPAIR);
    const float4* s4 = (const float4*)sOut;
#if USE_ASYNC
    #pragma unroll
    for (int k = 0; k < 8; ++k) {
      int i = tid + k * NTHREADS;
      __builtin_amdgcn_global_store_async_from_lds_b128(
          (gv4i*)(g + i), (sv4i*)(s4 + i), 0, 0);
    }
    __builtin_amdgcn_s_wait_asynccnt(0);
#else
    #pragma unroll
    for (int k = 0; k < 8; ++k)
      g[tid + k * NTHREADS] = s4[tid + k * NTHREADS];
#endif
  }
}

extern "C" void kernel_launch(void* const* d_in, const int* in_sizes, int n_in,
                              void* d_out, int out_size, void* d_ws, size_t ws_size,
                              hipStream_t stream) {
  const float* x  = (const float*)d_in[0];
  const float* dd = (const float*)d_in[1];
  const float* W1 = (const float*)d_in[2];
  const float* b1 = (const float*)d_in[3];
  const float* W2 = (const float*)d_in[4];
  const float* b2 = (const float*)d_in[5];
  const float* W3 = (const float*)d_in[6];
  const float* b3 = (const float*)d_in[7];
  float* outp = (float*)d_out;

  const int n    = in_sizes[0] / (ND * 12);   // 65536
  const int grid = n / BPW;                   // 4096 workgroups
  dijet_resnet_wmma<<<grid, NTHREADS, 0, stream>>>(x, dd, W1, b1, W2, b2, W3, b3, outp);
}
